// GQA_SA_41051297415425
// MI455X (gfx1250) — matrix operand order repaired and checked by reading, not compiled, not run on hardware
//
#include <hip/hip_runtime.h>
#include <hip/hip_bf16.h>

// ---------------------------------------------------------------------------
// GQA self-attention, CDNA5 (gfx1250) wave32 + WMMA f32_16x16x32_f16.
// B=2, S=1024, D=4096, Q_HEADS=32, KV_HEADS=8, D_HEAD=128, GROUPS=4.
// ASYNCcnt-tracked global->LDS async staging where no conversion is needed
// (flash-attn K tiles, f16 A tiles of the Wo GEMM).
// ---------------------------------------------------------------------------

typedef __attribute__((ext_vector_type(16))) _Float16 v16h;
typedef __attribute__((ext_vector_type(8)))  float    v8f;

union FragH {           // 16 halves = one WMMA A/B operand (8 VGPRs)
    v16h  v;
    uint4 u[2];
};

static __device__ __forceinline__ v8f wmma_f16(const v16h& a, const v16h& b, const v8f& c) {
    return __builtin_amdgcn_wmma_f32_16x16x32_f16(
        /*neg_a=*/false, a, /*neg_b=*/false, b,
        /*c_mod=*/(short)0, c, /*reuse_a=*/false, /*reuse_b=*/false);
}

// ---- async global->LDS staging (gfx1250 ASYNCcnt path), compile-safe gate --
#if defined(__has_builtin)
#  if __has_builtin(__builtin_amdgcn_global_load_async_to_lds_b128) && \
      __has_builtin(__builtin_amdgcn_s_wait_asynccnt)
#    define USE_ASYNC_LDS 1
#  endif
#endif
#ifndef USE_ASYNC_LDS
#  define USE_ASYNC_LDS 0
#endif

#if USE_ASYNC_LDS
// Builtin signature (from toolchain diagnostic): param0 = v4i in AS1 (global),
// param1 = v4i in AS3 (LDS), then imm offset, imm cpol.
typedef int v4i_vs __attribute__((vector_size(16)));
typedef __attribute__((address_space(1))) v4i_vs GV4;   // global int4
typedef __attribute__((address_space(3))) v4i_vs LV4;   // LDS int4
// 16-byte global->LDS async copy. AS1 numeric value == generic value for
// global pointers; AS3 offset == low 32 bits of the generic LDS address.
static __device__ __forceinline__ void async_copy16(const _Float16* g, _Float16* l) {
    __builtin_amdgcn_global_load_async_to_lds_b128(
        (GV4*)(unsigned long long)(uintptr_t)g,
        (LV4*)(unsigned int)(uintptr_t)l, 0, 0);
}
static __device__ __forceinline__ void async_fence() {
    __builtin_amdgcn_s_wait_asynccnt(0);
}
#else
static __device__ __forceinline__ void async_fence() {}
#endif

// ---- LDS staging helpers (fp32 or f16 source -> f16 LDS row chunk) --------
static __device__ __forceinline__ void stage16(const float* __restrict__ src,
                                               _Float16* __restrict__ dst) {
    const float4* s4 = reinterpret_cast<const float4*>(src);
#pragma unroll
    for (int j = 0; j < 4; ++j) {
        float4 f = s4[j];
        dst[4 * j + 0] = (_Float16)f.x;
        dst[4 * j + 1] = (_Float16)f.y;
        dst[4 * j + 2] = (_Float16)f.z;
        dst[4 * j + 3] = (_Float16)f.w;
    }
}
static __device__ __forceinline__ void stage16(const _Float16* __restrict__ src,
                                               _Float16* __restrict__ dst) {
#if USE_ASYNC_LDS
    async_copy16(src, dst);          // no conversion needed: async path,
    async_copy16(src + 8, dst + 8);  // bypasses VGPRs, tracked by ASYNCcnt
#else
    reinterpret_cast<uint4*>(dst)[0] = reinterpret_cast<const uint4*>(src)[0];
    reinterpret_cast<uint4*>(dst)[1] = reinterpret_cast<const uint4*>(src)[1];
#endif
}

// ---------------------------------------------------------------------------
// Generic tiled GEMM: C[M,N] = A[M,K] * W[K,N] + bias, fp32 out.
// Block tile 128x128, BK=32, 256 threads = 8 waves (4x2), wave tile 32x64.
// A staged [m][k] (A-frag layout: 2 contiguous 8-half chunks per lane).
// W staged transposed [n][k] (B-frag layout: 2 contiguous 8-half chunks per lane).
// ---------------------------------------------------------------------------
template <typename AT>
__global__ __launch_bounds__(256) void gemm_wmma(
    const AT* __restrict__ A, const float* __restrict__ W,
    const float* __restrict__ bias, float* __restrict__ C,
    int M, int N, int K) {
    constexpr int BM = 128, BN = 128, BK = 32;
    constexpr int LDA = BK + 8;   // halves; 80B row stride
    constexpr int LDB = BK + 8;
    __shared__ _Float16 lds_a[BM * LDA];
    __shared__ _Float16 lds_w[BN * LDB];

    const int tid  = threadIdx.x;
    const int lane = tid & 31;
    const int wid  = tid >> 5;
    const int bm   = blockIdx.x * BM;
    const int bn   = blockIdx.y * BN;
    const int wm   = (wid >> 1) * 32;   // 4 waves along M
    const int wn   = (wid & 1) * 64;    // 2 waves along N

    v8f c[2][4];
#pragma unroll
    for (int i = 0; i < 2; ++i)
#pragma unroll
        for (int j = 0; j < 4; ++j) c[i][j] = {};

    const int am = tid >> 1;              // A stage: row 0..127
    const int ak = (tid & 1) * 16;        // half of the 32-wide k strip
    const int wk = tid >> 3;              // W stage: k row 0..31
    const int wnn = (tid & 7) * 16;       // n strip 0..112

    for (int k0 = 0; k0 < K; k0 += BK) {
        // ---- stage A tile (async when AT==_Float16) ----
        stage16(A + (size_t)(bm + am) * K + k0 + ak, &lds_a[am * LDA + ak]);
        // ---- stage W tile (transposed into [n][k], fp32->f16) ----
        {
            const float* src = W + (size_t)(k0 + wk) * N + bn + wnn;
#pragma unroll
            for (int j = 0; j < 4; ++j) {
                float4 f = reinterpret_cast<const float4*>(src)[j];
                lds_w[(wnn + 4 * j + 0) * LDB + wk] = (_Float16)f.x;
                lds_w[(wnn + 4 * j + 1) * LDB + wk] = (_Float16)f.y;
                lds_w[(wnn + 4 * j + 2) * LDB + wk] = (_Float16)f.z;
                lds_w[(wnn + 4 * j + 3) * LDB + wk] = (_Float16)f.w;
            }
        }
        if (k0 + BK < K) {   // pull next tile toward the caches
            __builtin_prefetch(A + (size_t)(bm + am) * K + k0 + BK + ak, 0, 0);
            __builtin_prefetch(W + (size_t)(k0 + BK + wk) * N + bn + wnn, 0, 0);
        }
        async_fence();
        __syncthreads();

        // ---- fragments + WMMA ----
        FragH a[2];
        {
            const int mrow = wm + (lane & 15);
            const int kb   = (lane >> 4) * 8;
#pragma unroll
            for (int mi = 0; mi < 2; ++mi) {
                const _Float16* p = &lds_a[(mrow + mi * 16) * LDA + kb];
                a[mi].u[0] = *reinterpret_cast<const uint4*>(p);
                a[mi].u[1] = *reinterpret_cast<const uint4*>(p + 16);
            }
        }
        FragH b[4];
        {
            const int ncol = wn + (lane & 15);
            const int kb   = (lane >> 4) * 8;
#pragma unroll
            for (int ni = 0; ni < 4; ++ni) {
                const _Float16* p = &lds_w[(ncol + ni * 16) * LDB + kb];
                b[ni].u[0] = *reinterpret_cast<const uint4*>(p);
                b[ni].u[1] = *reinterpret_cast<const uint4*>(p + 16);
            }
        }
#pragma unroll
        for (int mi = 0; mi < 2; ++mi)
#pragma unroll
            for (int ni = 0; ni < 4; ++ni)
                c[mi][ni] = wmma_f16(a[mi].v, b[ni].v, c[mi][ni]);
        __syncthreads();
    }

    // ---- epilogue: bias + fp32 store (C layout: M=(lane>>4)*8+i, N=lane&15)
    const int ncol0 = bn + wn + (lane & 15);
    const int mrow0 = bm + wm + (lane >> 4) * 8;
#pragma unroll
    for (int ni = 0; ni < 4; ++ni) {
        const int n  = ncol0 + ni * 16;
        const float bi = bias[n];
#pragma unroll
        for (int mi = 0; mi < 2; ++mi)
#pragma unroll
            for (int i = 0; i < 8; ++i)
                C[(size_t)(mrow0 + mi * 16 + i) * N + n] = c[mi][ni][i] + bi;
    }
}

// ---------------------------------------------------------------------------
// RoPE (optional) + fp32 -> f16 + transpose (b,s,h,dh) -> (b,h,s,dh).
// One 64-thread block per (s, h, b); thread i owns pair (2i, 2i+1).
// ---------------------------------------------------------------------------
__global__ __launch_bounds__(64) void rope_convert(
    const float* __restrict__ in, const float* __restrict__ freqs,
    _Float16* __restrict__ out, int H, int apply_rope) {
    constexpr int S = 1024, DH = 128;
    const int s = blockIdx.x, h = blockIdx.y, b = blockIdx.z;
    const int i = threadIdx.x;   // pair index 0..63
    const float* row = in + ((size_t)(b * S + s) * H + h) * DH;
    float xr = row[2 * i], xi = row[2 * i + 1];
    float o0, o1;
    if (apply_rope) {
        const float ang = freqs[s * (DH / 2) + i];
        const float cs = __cosf(ang), sn = __sinf(ang);
        o0 = xr * cs - xi * sn;
        o1 = xr * sn + xi * cs;
    } else {
        o0 = xr; o1 = xi;
    }
    _Float16* orow = out + ((size_t)(b * H + h) * S + s) * DH;
    orow[2 * i] = (_Float16)o0;
    orow[2 * i + 1] = (_Float16)o1;
}

// ---------------------------------------------------------------------------
// Flash attention, causal, GQA (kv = hq % 8). Block = 8 waves, 128 q rows
// (16 per wave). Key tiles of 64; K in LDS as [t][dh] (async-staged),
// V transposed [dh][t] (VGPR scatter). Output written f16 into the
// reference's scrambled (b,kv,g,s,dh)->(b,s,d) view so Wo is a plain GEMM.
// ---------------------------------------------------------------------------
__global__ __launch_bounds__(256) void flash_attn(
    const _Float16* __restrict__ Qh, const _Float16* __restrict__ Kh,
    const _Float16* __restrict__ Vh, _Float16* __restrict__ scr) {
    constexpr int S = 1024, DH = 128, TK = 64;
    constexpr int LDK = DH + 8;   // 136 halves
    constexpr int LDV = TK + 8;   // 72
    constexpr int LDP = TK + 8;   // 72
    __shared__ _Float16 lds_k[TK * LDK];
    __shared__ _Float16 lds_v[DH * LDV];
    __shared__ _Float16 lds_p[8 * 16 * LDP];

    const int tid = threadIdx.x, lane = tid & 31, wid = tid >> 5;
    const int qt = blockIdx.x;          // 128-row q tile
    const int hq = blockIdx.y;          // query head
    const int b  = blockIdx.z;
    const int kv = hq & 7;              // reference: kv = hq % KV_HEADS
    const int qbase = qt * 128 + wid * 16;
    const float scale = 0.08838834764831845f;   // 1/sqrt(128)

    // Q A-fragments (4 k-steps of 32) straight from global, kept in VGPRs.
    FragH qf[4];
    {
        const _Float16* qrow =
            Qh + ((size_t)(b * 32 + hq) * S + qbase + (lane & 15)) * DH;
        const int kb = (lane >> 4) * 8;
#pragma unroll
        for (int j = 0; j < 4; ++j) {
            qf[j].u[0] = *reinterpret_cast<const uint4*>(qrow + j * 32 + kb);
            qf[j].u[1] = *reinterpret_cast<const uint4*>(qrow + j * 32 + kb + 16);
        }
    }

    v8f o[8];
#pragma unroll
    for (int d = 0; d < 8; ++d) o[d] = {};
    float rm[8], rl[8];
#pragma unroll
    for (int i = 0; i < 8; ++i) { rm[i] = -1e30f; rl[i] = 0.f; }

    const int ntiles = (qt + 1) * 2;    // causal: keys 0 .. (qt+1)*128-1
    for (int kt = 0; kt < ntiles; ++kt) {
        const int t0 = kt * TK;
        __syncthreads();
        // ---- stage K [t][dh] (async) and V transposed [dh][t] ----
        {
            const _Float16* ksrc = Kh + ((size_t)(b * 8 + kv) * S + t0) * DH;
            const _Float16* vsrc = Vh + ((size_t)(b * 8 + kv) * S + t0) * DH;
#pragma unroll
            for (int c4 = 0; c4 < 4; ++c4) {
                const int idx = (tid * 4 + c4) * 8;   // half index in 64x128
                const int tr = idx >> 7;
                const int dh = idx & 127;
#if USE_ASYNC_LDS
                async_copy16(ksrc + tr * DH + dh, &lds_k[tr * LDK + dh]);
#else
                *reinterpret_cast<uint4*>(&lds_k[tr * LDK + dh]) =
                    *reinterpret_cast<const uint4*>(ksrc + tr * DH + dh);
#endif
                uint4 raw = *reinterpret_cast<const uint4*>(vsrc + tr * DH + dh);
                const _Float16* hp = reinterpret_cast<const _Float16*>(&raw);
#pragma unroll
                for (int e = 0; e < 8; ++e)
                    lds_v[(dh + e) * LDV + tr] = hp[e];
            }
        }
        async_fence();
        __syncthreads();

        if (t0 <= qbase) {   // wave-uniform: tile has unmasked work
            // ---- scores = Q @ K^T (4 n-tiles x 4 k-steps) ----
            v8f sc[4];
#pragma unroll
            for (int ni = 0; ni < 4; ++ni) sc[ni] = {};
            {
                const int kbB = (lane >> 4) * 8;
#pragma unroll
                for (int ni = 0; ni < 4; ++ni) {
                    const int tcol = ni * 16 + (lane & 15);
#pragma unroll
                    for (int j = 0; j < 4; ++j) {
                        FragH bk;
                        const _Float16* p = &lds_k[tcol * LDK + j * 32 + kbB];
                        bk.u[0] = *reinterpret_cast<const uint4*>(p);
                        bk.u[1] = *reinterpret_cast<const uint4*>(p + 16);
                        sc[ni] = wmma_f16(qf[j].v, bk.v, sc[ni]);
                    }
                }
            }
            // ---- scale + causal mask + tile row-max ----
            float tmax[8];
#pragma unroll
            for (int i = 0; i < 8; ++i) tmax[i] = -1e30f;
            const int myrow = qbase + (lane >> 4) * 8;
            const int mycol = t0 + (lane & 15);
#pragma unroll
            for (int ni = 0; ni < 4; ++ni)
#pragma unroll
                for (int i = 0; i < 8; ++i) {
                    float v = sc[ni][i] * scale;
                    if (mycol + ni * 16 > myrow + i) v = -1e30f;
                    sc[ni][i] = v;
                    tmax[i] = fmaxf(tmax[i], v);
                }
#pragma unroll
            for (int i = 0; i < 8; ++i)
#pragma unroll
                for (int m = 1; m < 16; m <<= 1)
                    tmax[i] = fmaxf(tmax[i], __shfl_xor(tmax[i], m, 32));
            // ---- online softmax update ----
            float alpha[8], tsum[8];
#pragma unroll
            for (int i = 0; i < 8; ++i) {
                const float nm = fmaxf(rm[i], tmax[i]);
                alpha[i] = __expf(rm[i] - nm);
                rm[i] = nm;
                tsum[i] = 0.f;
            }
#pragma unroll
            for (int ni = 0; ni < 4; ++ni)
#pragma unroll
                for (int i = 0; i < 8; ++i) {
                    const float p = __expf(sc[ni][i] - rm[i]);
                    sc[ni][i] = p;
                    tsum[i] += p;
                }
#pragma unroll
            for (int i = 0; i < 8; ++i) {
#pragma unroll
                for (int m = 1; m < 16; m <<= 1)
                    tsum[i] += __shfl_xor(tsum[i], m, 32);
                rl[i] = rl[i] * alpha[i] + tsum[i];
            }
#pragma unroll
            for (int d = 0; d < 8; ++d)
#pragma unroll
                for (int i = 0; i < 8; ++i) o[d][i] *= alpha[i];

            // ---- P (C layout) -> per-wave LDS -> A layout ----
            _Float16* pw = &lds_p[wid * 16 * LDP];
#pragma unroll
            for (int ni = 0; ni < 4; ++ni)
#pragma unroll
                for (int i = 0; i < 8; ++i)
                    pw[((lane >> 4) * 8 + i) * LDP + ni * 16 + (lane & 15)] =
                        (_Float16)sc[ni][i];
            FragH pf[2];
            {
                const int kb = (lane >> 4) * 8;
#pragma unroll
                for (int j = 0; j < 2; ++j) {
                    const _Float16* p = &pw[(lane & 15) * LDP + j * 32 + kb];
                    pf[j].u[0] = *reinterpret_cast<const uint4*>(p);
                    pf[j].u[1] = *reinterpret_cast<const uint4*>(p + 16);
                }
            }
            // ---- O += P @ V ----
            {
                const int kbB = (lane >> 4) * 8;
#pragma unroll
                for (int nd = 0; nd < 8; ++nd) {
                    const int dh = nd * 16 + (lane & 15);
#pragma unroll
                    for (int j = 0; j < 2; ++j) {
                        FragH bv;
                        const _Float16* p = &lds_v[dh * LDV + j * 32 + kbB];
                        bv.u[0] = *reinterpret_cast<const uint4*>(p);
                        bv.u[1] = *reinterpret_cast<const uint4*>(p + 16);
                        o[nd] = wmma_f16(pf[j].v, bv.v, o[nd]);
                    }
                }
            }
        }
    }

    // ---- normalize + store into scrambled layout ----
    float inv[8];
#pragma unroll
    for (int i = 0; i < 8; ++i) inv[i] = 1.f / rl[i];
    const int hscr = (hq & 7) * 4 + (hq >> 3);   // h' = kv*GROUPS + g
#pragma unroll
    for (int nd = 0; nd < 8; ++nd) {
        const int dh = nd * 16 + (lane & 15);
#pragma unroll
        for (int i = 0; i < 8; ++i) {
            const int srow = qbase + (lane >> 4) * 8 + i;
            const size_t idx = ((size_t)b * 1024 + hscr * 32 + (srow >> 5)) * 4096 +
                               (size_t)(srow & 31) * 128 + dh;
            scr[idx] = (_Float16)(o[nd][i] * inv[i]);
        }
    }
}

// ---------------------------------------------------------------------------
extern "C" void kernel_launch(void* const* d_in, const int* in_sizes, int n_in,
                              void* d_out, int out_size, void* d_ws, size_t ws_size,
                              hipStream_t stream) {
    (void)in_sizes; (void)n_in; (void)out_size; (void)ws_size;
    const float* x    = (const float*)d_in[0];
    const float* rope = (const float*)d_in[1];
    // d_in[2] = causal mask (implemented analytically)
    const float* Wq = (const float*)d_in[3];
    const float* bq = (const float*)d_in[4];
    const float* Wk = (const float*)d_in[5];
    const float* bk = (const float*)d_in[6];
    const float* Wv = (const float*)d_in[7];
    const float* bv = (const float*)d_in[8];
    const float* Wo = (const float*)d_in[9];
    const float* bo = (const float*)d_in[10];
    float* out = (float*)d_out;

    // Workspace map (peak ~72 MB):
    //   [0,  32MB)  qf32   (later reused as f16 scrambled attn buffer)
    //   [32, 40MB)  kf32
    //   [40, 48MB)  vf32
    //   [48, 64MB)  Qh f16 head-major
    //   [64, 68MB)  Kh f16 head-major
    //   [68, 72MB)  Vh f16 head-major
    char* ws = (char*)d_ws;
    float*     qf32 = (float*)(ws);
    float*     kf32 = (float*)(ws + (size_t)33554432);
    float*     vf32 = (float*)(ws + (size_t)41943040);
    _Float16*  Qh   = (_Float16*)(ws + (size_t)50331648);
    _Float16*  Kh   = (_Float16*)(ws + (size_t)67108864);
    _Float16*  Vh   = (_Float16*)(ws + (size_t)71303168);
    _Float16*  scr  = (_Float16*)(ws);   // reuse region A after RoPE pass

    const int M = 2048;   // B*S
    // 1-3) QKV projections (fp32 out, pre-RoPE)
    gemm_wmma<float><<<dim3(16, 32), 256, 0, stream>>>(x, Wq, bq, qf32, M, 4096, 4096);
    gemm_wmma<float><<<dim3(16, 8),  256, 0, stream>>>(x, Wk, bk, kf32, M, 1024, 4096);
    gemm_wmma<float><<<dim3(16, 8),  256, 0, stream>>>(x, Wv, bv, vf32, M, 1024, 4096);
    // 4-6) RoPE (Q,K) / convert (V) -> f16 head-major
    rope_convert<<<dim3(1024, 32, 2), 64, 0, stream>>>(qf32, rope, Qh, 32, 1);
    rope_convert<<<dim3(1024, 8, 2),  64, 0, stream>>>(kf32, rope, Kh, 8, 1);
    rope_convert<<<dim3(1024, 8, 2),  64, 0, stream>>>(vf32, rope, Vh, 8, 0);
    // 7) causal flash attention -> scrambled f16 buffer
    flash_attn<<<dim3(8, 32, 2), 256, 0, stream>>>(Qh, Kh, Vh, scr);
    // 8) output projection
    gemm_wmma<_Float16><<<dim3(16, 32), 256, 0, stream>>>(scr, Wo, bo, out, M, 4096, 4096);
}
